// Word2MatEncoder_72962904425072
// MI455X (gfx1250) — matrix operand order repaired and checked by reading, not compiled, not run on hardware
//
#include <hip/hip_runtime.h>

// CMOW word-matrix chain: out[b] = table[sent[b,0]] @ ... @ table[sent[b,63]] (28x28)
// One workgroup (128 threads / 4 waves) per batch item; 28x28 zero-padded to 32x32.
// Per chain step: 4 waves each do one v_wmma_f32_16x16x32_bf16 (K=32 covers padded dim).
// The per-step table-row gather is offloaded to the Tensor Data Mover:
// tensor_load_to_lds in iterate mode copies the 28x28 f32 matrix TRANSPOSED
// (column-major) into LDS while the waves compute the current step (TENSORcnt overlap).

typedef __attribute__((ext_vector_type(16))) __bf16       v16bf;
typedef __attribute__((ext_vector_type(8)))  float        v8f;
typedef __attribute__((ext_vector_type(4)))  unsigned int vu4;
typedef __attribute__((ext_vector_type(4)))  unsigned int u32x4;
typedef __attribute__((ext_vector_type(8)))  int          i32x8;
typedef __attribute__((ext_vector_type(4)))  int          i32x4;

#define MAT 28
#define PAD 32
#define SEQL 64
#define EMB 784

union V16 { v16bf v; vu4 q[2]; };

// TDM: copy table row (28x28 f32, row-major in memory) into LDS as column-major
// 32-float-stride tile. Iterate mode: each iteration moves one source column
// (tile 1 x 28, stride 28) to a contiguous 28-float LDS run; lds_addr += 32 floats,
// global_addr += 1 float per iteration, 28 iterations.
__device__ __forceinline__ void tdm_load_col_major(unsigned lds_off, const float* gsrc) {
    unsigned long long ga = (unsigned long long)gsrc;

    u32x4 g0;
    g0[0] = 1u;                                               // count=1 (valid user D#)
    g0[1] = lds_off;                                          // lds_addr (bytes)
    g0[2] = (unsigned)(ga & 0xFFFFFFFFull);                   // global_addr[31:0]
    g0[3] = (unsigned)((ga >> 32) & 0x1FFFFFFull)             // global_addr[56:32]
          | (2u << 30);                                       // type=2 ("image")

    i32x8 g1;
    g1[0] = (2 << 16) | (1 << 19);       // data_size=4B, iterate_enable=1, no multicast
    g1[1] = (MAT & 0xFFFF) << 16;        // tensor_dim0[15:0] = 28   (bits 79:48)
    g1[2] = ((MAT >> 16) & 0xFFFF) | ((MAT & 0xFFFF) << 16);  // dim0 hi | tensor_dim1 lo = 28
    g1[3] = ((MAT >> 16) & 0xFFFF) | (1 << 16);               // dim1 hi | tile_dim0 = 1
    g1[4] = MAT;                         // tile_dim1 = 28, tile_dim2 = 0
    g1[5] = MAT;                         // tensor_dim0_stride[31:0] = 28
    g1[6] = 0;                           // stride hi | tensor_dim1_stride lo
    g1[7] = 0;

    i32x4 g2;
    g2[0] = 0;                           // tensor_dim2 (unused)
    g2[1] = PAD;                         // lds_addr_increment = 32 elements (col stride)
    g2[2] = 1;                           // global_addr_increment = 1 element
    g2[3] = (MAT - 1) << 16;             // iterate_count = 27 -> 28 iterations

    i32x4 g3 = {0, 0, 0, 0};
    i32x8 g4 = {0, 0, 0, 0, 0, 0, 0, 0}; // 6-arg toolchain form: extra descriptor group

    __builtin_amdgcn_tensor_load_to_lds(g0, g1, g2, g3, g4, /*cpol=*/0);
}

__global__ __launch_bounds__(128)
void cmow_chain_kernel(const int* __restrict__ sent,
                       const float* __restrict__ table,
                       float* __restrict__ out)
{
    __shared__ __bf16 prod[2][PAD * PAD];   // running product, row-major bf16
    __shared__ float  matF[2][PAD * PAD];   // next matrix, column-major f32 (TDM dest)
    __shared__ int    idxs[SEQL];

    const int b    = blockIdx.x;
    const int tid  = threadIdx.x;
    const int lane = tid & 31;
    const int wave = tid >> 5;
    const int l    = lane & 15;        // lane-within-half
    const int half = lane >> 4;        // 0: lanes 0-15, 1: lanes 16-31
    const int ti   = (wave >> 1) * 16; // C-tile row base
    const int tj   = (wave & 1) * 16;  // C-tile col base

    // Zero LDS once: establishes zero padding (rows/cols 28..31 never rewritten).
    for (int e = tid; e < PAD * PAD; e += 128) {
        prod[0][e] = (__bf16)0.0f; prod[1][e] = (__bf16)0.0f;
        matF[0][e] = 0.0f;         matF[1][e] = 0.0f;
    }
    if (tid < SEQL) idxs[tid] = sent[b * SEQL + tid];
    __syncthreads();

    // Kick off DMA of M1 (col-major into matF[0]) while loading P = M0 (bf16, row-major).
    if (wave == 0) {
        unsigned lds_off = (unsigned)(unsigned long long)(const void*)&matF[0][0];
        tdm_load_col_major(lds_off, table + (long)idxs[1] * EMB);
    }
    {
        const float* r0 = table + (long)idxs[0] * EMB;
        for (int e = tid; e < EMB; e += 128) {
            int r = e / MAT, c = e - r * MAT;
            prod[0][r * PAD + c] = (__bf16)r0[e];
        }
    }
    __builtin_amdgcn_s_wait_tensorcnt(0);
    __syncthreads();

    v8f acc = {};
    int cur = 0;
    for (int s = 1; s < SEQL; ++s) {
        const int nxt = cur ^ 1;

        // Issue DMA for step s+1 first: overlaps with this step's WMMA work.
        if (s + 1 < SEQL && wave == 0) {
            unsigned lds_off = (unsigned)(unsigned long long)(const void*)&matF[nxt][0];
            tdm_load_col_major(lds_off, table + (long)idxs[s + 1] * EMB);
            if (s + 2 < SEQL) { // warm L2 for the DMA after this one
                const char* pf = (const char*)(table + (long)idxs[s + 2] * EMB);
                if (lane < 25) __builtin_prefetch(pf + lane * 128, 0, 1); // 25*128B >= 3136B
            }
        }

        // A operand (16x32 bf16, rows ti..ti+15 of running product).
        // 16-bit A layout: lanes 0-15 row M=ti+l hold K=0..7 then 16..23;
        // lanes 16-31 hold K=8..15 then 24..31 -> two contiguous 16B chunks.
        V16 A;
        {
            const int kbase = half * 8;
            const vu4* pa = (const vu4*)&prod[cur][(ti + l) * PAD + kbase];
            A.q[0] = pa[0]; // K = kbase .. kbase+7
            A.q[1] = pa[2]; // K = kbase+16 .. kbase+23
        }

        // B operand (32x16, cols tj..tj+15): col-major f32 in LDS -> each lane
        // reads K=half*8..+7 then K=16+half*8..+7 (same order as A); convert to bf16 on the fly.
        v16bf Bv;
        {
            const float* Mc = &matF[cur][(tj + l) * PAD + half * 8];
            #pragma unroll
            for (int e = 0; e < 16; ++e) Bv[e] = (__bf16)Mc[e + (e >= 8 ? 8 : 0)];
        }

        // D = A x B (+0): fresh product each step, f32 accumulators.
        v8f cz = {};
        acc = __builtin_amdgcn_wmma_f32_16x16x32_bf16(
            /*neg_a=*/false, A.v, /*neg_b=*/false, Bv,
            /*c_mod=*/(short)0, cz, /*reuse_a=*/false, /*reuse_b=*/false);

        // Write D back as bf16 into the other product buffer.
        // C/D layout: VGPR v -> row ti + half*8 + v, col tj + l.
        {
            const int coln = tj + l;
            #pragma unroll
            for (int v = 0; v < 8; ++v)
                prod[nxt][(ti + half * 8 + v) * PAD + coln] = (__bf16)acc[v];
        }

        // DMA must have landed before anyone reads matF[nxt] next step.
        __builtin_amdgcn_s_wait_tensorcnt(0);
        __syncthreads();
        cur = nxt;
    }

    // Final product tile is still in f32 accumulators: write the 28x28 region.
    {
        const int coln = tj + l;
        #pragma unroll
        for (int v = 0; v < 8; ++v) {
            const int rown = ti + half * 8 + v;
            if (rown < MAT && coln < MAT)
                out[(long)b * EMB + rown * MAT + coln] = acc[v];
        }
    }
}

extern "C" void kernel_launch(void* const* d_in, const int* in_sizes, int n_in,
                              void* d_out, int out_size, void* d_ws, size_t ws_size,
                              hipStream_t stream) {
    (void)n_in; (void)out_size; (void)d_ws; (void)ws_size;
    const int*   sent  = (const int*)d_in[0];
    const float* table = (const float*)d_in[1];
    float*       out   = (float*)d_out;
    const int batch = in_sizes[0] / SEQL; // 1024
    cmow_chain_kernel<<<batch, 128, 0, stream>>>(sent, table, out);
}
